// ContrastiveLossELI5_29334626631753
// MI455X (gfx1250) — matrix operand order repaired and checked by reading, not compiled, not run on hardware
//
#include <hip/hip_runtime.h>
#include <hip/hip_bf16.h>

// ---------------------------------------------------------------------------
// NT-Xent (SimCLR) loss for B=4096, D=256 on MI455X (gfx1250, wave32).
//  sim = Z Z^T via V_WMMA_F32_16X16X32_BF16, fused exp/row-sum epilogue.
//  - 8 waves of a block share one column segment (WGP$ B-tile reuse).
//  - Two column tiles per iteration share the register-resident A fragments
//    (two independent WMMA chains per K-chunk -> better load/WMMA overlap).
//  - Hot path is branch-free on the denominator accumulator; diagonal and
//    positive-pair values are extracted in rare scalar-branch blocks and
//    corrected after the reduction.
// ---------------------------------------------------------------------------

typedef __attribute__((ext_vector_type(16))) __bf16 v16bf;
typedef __attribute__((ext_vector_type(8)))  __bf16 v8bf;
typedef __attribute__((ext_vector_type(8)))  float  v8f;

#define NROWS   8192      // 2N
#define DIM     256
#define HALFN   4096      // N (power of two -> pos partner = row ^ HALFN)
#define INV_T   2.0f      // 1 / TEMPERATURE
#define NSEG    8         // column segments per row-group
#define COLSEG  (NROWS / NSEG)     // 1024 columns per segment
#define NTILES  (COLSEG / 16)      // 64 column tiles per segment

union ABu { v16bf v; v8bf h[2]; };

// --- kernel 0: zero the accumulators and the output scalar ------------------
__global__ void k_zero(float* __restrict__ acc, float* __restrict__ out) {
    int tid = blockIdx.x * blockDim.x + threadIdx.x;
    if (tid < 2 * NROWS) acc[tid] = 0.0f;
    if (tid == 0) out[0] = 0.0f;
}

// --- kernel 1: row-normalize fp32 inputs, emit bf16 Z -----------------------
__global__ void __launch_bounds__(256)
k_normalize(const float* __restrict__ emb_i, const float* __restrict__ emb_j,
            __bf16* __restrict__ reps) {
    const int lane = threadIdx.x & 31;
    const int row  = blockIdx.x * 8 + (threadIdx.x >> 5);   // one wave per row
    const float* src = (row < HALFN) ? (emb_i + (size_t)row * DIM)
                                     : (emb_j + (size_t)(row - HALFN) * DIM);
    const float4 x0 = *(const float4*)(src + lane * 8);
    const float4 x1 = *(const float4*)(src + lane * 8 + 4);
    float ss = x0.x*x0.x + x0.y*x0.y + x0.z*x0.z + x0.w*x0.w
             + x1.x*x1.x + x1.y*x1.y + x1.z*x1.z + x1.w*x1.w;
    #pragma unroll
    for (int m = 1; m <= 16; m <<= 1) ss += __shfl_xor(ss, m, 32);
    const float scale = 1.0f / fmaxf(sqrtf(ss), 1e-12f);

    v8bf z;
    z[0] = (__bf16)(x0.x * scale); z[1] = (__bf16)(x0.y * scale);
    z[2] = (__bf16)(x0.z * scale); z[3] = (__bf16)(x0.w * scale);
    z[4] = (__bf16)(x1.x * scale); z[5] = (__bf16)(x1.y * scale);
    z[6] = (__bf16)(x1.z * scale); z[7] = (__bf16)(x1.w * scale);
    *(v8bf*)(reps + (size_t)row * DIM + lane * 8) = z;
}

// --- kernel 2: Gram matrix tiles via WMMA + fused exp row-sums --------------
// Block b: segment sg = b % 8 (shared by all 8 waves -> WGP$ B-tile reuse),
//          row-groups (b/8)*8 + waveId (rows rg*16 .. rg*16+15).
__global__ void __launch_bounds__(256)
k_simloss(const __bf16* __restrict__ reps,
          float* __restrict__ denom, float* __restrict__ possum) {
    const int lane    = threadIdx.x & 31;
    const int sg      = blockIdx.x & (NSEG - 1);
    const int rg      = (blockIdx.x >> 3) * 8 + (threadIdx.x >> 5);
    // rg/r0 are wave-uniform by construction; readfirstlane makes that
    // provable to the compiler -> scalar (s_cmp/s_cbranch) tile checks.
    const int r0      = __builtin_amdgcn_readfirstlane(rg * 16);
    const int l15     = lane & 15;
    const int halfSel = lane >> 4;           // 0: lanes 0-15, 1: lanes 16-31
    const int grBase  = r0 + (halfSel << 3); // row of C VGPR v is grBase + v

    // ---- A fragments: 16 rows x 256 K, register-resident (64 VGPRs) ----
    // A k order used here (contiguous, matches B): lanes 0-15 hold K {c*32+0..15},
    //                  lanes 16-31 hold K {c*32+16..31}.
    const __bf16* arp = reps + (size_t)(r0 + l15) * DIM + halfSel * 16;
    v16bf a[8];
    #pragma unroll
    for (int c = 0; c < 8; ++c) {
        ABu u;
        u.h[0] = *(const v8bf*)(arp + c * 32);
        u.h[1] = *(const v8bf*)(arp + c * 32 + 8);
        a[c] = u.v;
    }

    float dacc[8] = {0.f,0.f,0.f,0.f,0.f,0.f,0.f,0.f};  // sum exp (ALL cols)
    float dval[8] = {0.f,0.f,0.f,0.f,0.f,0.f,0.f,0.f};  // raw diagonal sim
    float pval[8] = {0.f,0.f,0.f,0.f,0.f,0.f,0.f,0.f};  // raw positive sim

    const int jbase   = sg * COLSEG;         // scalar (from blockIdx)
    const int posTile = r0 ^ HALFN;          // scalar: tile holding positives
    for (int t = 0; t < NTILES; t += 2) {
        const int j0a = jbase + t * 16;      // scalar
        const int j0b = j0a + 16;            // scalar
        const int gca = j0a + l15;           // lane's column in tile a
        const int gcb = gca + 16;            // lane's column in tile b
        // B k order used here (contiguous, matches A): lanes 0-15 -> N=lane, K=0..15
        //                                 lanes 16-31 -> N=lane-16, K=16..31
        const __bf16* brpA = reps + (size_t)gca * DIM + halfSel * 16;
        const __bf16* brpB = brpA + 16 * DIM;
        if (t + 2 < NTILES) {                // prefetch next pair's B rows
            __builtin_prefetch(brpA + 32 * DIM, 0, 1);
            __builtin_prefetch(brpB + 32 * DIM, 0, 1);
        }

        v8f cf0 = {}, cf1 = {};
        #pragma unroll
        for (int c = 0; c < 8; ++c) {
            const v16bf b0 = *(const v16bf*)(brpA + c * 32);
            const v16bf b1 = *(const v16bf*)(brpB + c * 32);
            cf0 = __builtin_amdgcn_wmma_f32_16x16x32_bf16(
                      false, a[c], false, b0, (short)0, cf0, false, false);
            cf1 = __builtin_amdgcn_wmma_f32_16x16x32_bf16(
                      false, a[c], false, b1, (short)0, cf1, false, false);
        }

        // Hot path: unconditional exp accumulation (includes the diagonal;
        // corrected after the reduction). Co-executes under the WMMA pipe.
        #pragma unroll
        for (int v = 0; v < 8; ++v) {
            dacc[v] += __expf(cf0[v] * INV_T);
            dacc[v] += __expf(cf1[v] * INV_T);
        }

        // Rare tiles (scalar branches, EXEC untouched): extract raw values
        // without touching dacc, so no phi-forking of the hot accumulator.
        if (j0a == r0) {
            #pragma unroll
            for (int v = 0; v < 8; ++v)
                dval[v] += (gca == grBase + v) ? cf0[v] : 0.0f;
        } else if (j0a == posTile) {
            #pragma unroll
            for (int v = 0; v < 8; ++v)
                pval[v] += (gca == ((grBase + v) ^ HALFN)) ? cf0[v] : 0.0f;
        }
        if (j0b == r0) {
            #pragma unroll
            for (int v = 0; v < 8; ++v)
                dval[v] += (gcb == grBase + v) ? cf1[v] : 0.0f;
        } else if (j0b == posTile) {
            #pragma unroll
            for (int v = 0; v < 8; ++v)
                pval[v] += (gcb == ((grBase + v) ^ HALFN)) ? cf1[v] : 0.0f;
        }
    }

    // reduce within each 16-lane half (rows grBase+0 .. grBase+7)
    #pragma unroll
    for (int m = 1; m <= 8; m <<= 1) {
        #pragma unroll
        for (int v = 0; v < 8; ++v) {
            dacc[v] += __shfl_xor(dacc[v], m, 32);
            dval[v] += __shfl_xor(dval[v], m, 32);
            pval[v] += __shfl_xor(pval[v], m, 32);
        }
    }
    // subtract exp(diagonal) once: only the wave whose segment contains the
    // diagonal tile has dval != 0 (scalar-guarded for exactness).
    if (r0 >= jbase && r0 < jbase + COLSEG) {
        #pragma unroll
        for (int v = 0; v < 8; ++v)
            dacc[v] -= __expf(dval[v] * INV_T);
    }
    if (l15 == 0) {
        #pragma unroll
        for (int v = 0; v < 8; ++v) {
            atomicAdd(&denom[grBase + v],  dacc[v]);
            atomicAdd(&possum[grBase + v], pval[v]);
        }
    }
}

// --- kernel 3: loss = (1/2N) * sum_i [ pos_i/T - log(denom_i) ] -------------
__global__ void k_finalize(const float* __restrict__ denom,
                           const float* __restrict__ possum,
                           float* __restrict__ out) {
    const int i = blockIdx.x * blockDim.x + threadIdx.x;   // 8192 threads
    float term = (possum[i] * INV_T - __logf(denom[i])) * (1.0f / (float)NROWS);
    #pragma unroll
    for (int m = 1; m <= 16; m <<= 1) term += __shfl_xor(term, m, 32);
    if ((threadIdx.x & 31) == 0) atomicAdd(out, term);
}

extern "C" void kernel_launch(void* const* d_in, const int* in_sizes, int n_in,
                              void* d_out, int out_size, void* d_ws, size_t ws_size,
                              hipStream_t stream) {
    (void)in_sizes; (void)n_in; (void)out_size; (void)ws_size;
    const float* emb_i = (const float*)d_in[0];
    const float* emb_j = (const float*)d_in[1];
    float* out = (float*)d_out;

    // workspace layout: [0, 4 MB) bf16 Z ; then denom[8192] f32 ; pos[8192] f32
    __bf16* reps  = (__bf16*)d_ws;
    float* denom  = (float*)((char*)d_ws + (size_t)NROWS * DIM * sizeof(__bf16));
    float* possum = denom + NROWS;

    k_zero     <<<(2 * NROWS + 255) / 256, 256, 0, stream>>>(denom, out);
    k_normalize<<<NROWS / 8,               256, 0, stream>>>(emb_i, emb_j, reps);
    k_simloss  <<<(NROWS / 16) * NSEG / 8, 256, 0, stream>>>(reps, denom, possum);
    k_finalize <<<NROWS / 256,             256, 0, stream>>>(denom, possum, out);
}
